// KernelHerding_56487409877229
// MI455X (gfx1250) — matrix operand order repaired and checked by reading, not compiled, not run on hardware
//
#include <hip/hip_runtime.h>
#include <hip/hip_bf16.h>

#define NROWS 16384
#define DIMK  128
// exp(-0.5*x) == exp2(-0.5*log2(e)*x)
#define NEG_HALF_LOG2E (-0.7213475204444817f)
// LDS row stride for a 16x128 bf16 tile: 256B data + 16B pad (bank de-conflict)
#define LDS_STRIDE 272

typedef __bf16 bf16_t;
typedef __attribute__((ext_vector_type(16))) __bf16 v16bf;
typedef __attribute__((ext_vector_type(8)))  __bf16 v8bf;
typedef __attribute__((ext_vector_type(4)))  __bf16 v4bf;
typedef __attribute__((ext_vector_type(8)))  float  v8f;

// exp2 via the hardware transcendental unit: one v_exp_f32.
__device__ __forceinline__ float fast_exp2(float x) {
    return __builtin_amdgcn_exp2f(x);
}

// ---------------------------------------------------------------------------
// K1: per-row squared norm (f32) + bf16 conversion of x. One wave per row.
// ---------------------------------------------------------------------------
__global__ void k_prep(const float* __restrict__ x, bf16_t* __restrict__ xb,
                       float* __restrict__ sq) {
    const int wave = (blockIdx.x * blockDim.x + threadIdx.x) >> 5;
    const int lane = threadIdx.x & 31;
    const int row  = wave;                       // 16384 waves
    const float4* xr = (const float4*)(x + (size_t)row * DIMK);
    float4 v = xr[lane];                         // 32 lanes * 4 = 128
    float s = v.x * v.x + v.y * v.y + v.z * v.z + v.w * v.w;

    v4bf o;
    o[0] = (__bf16)v.x; o[1] = (__bf16)v.y; o[2] = (__bf16)v.z; o[3] = (__bf16)v.w;
    *(v4bf*)(xb + (size_t)row * DIMK + lane * 4) = o;   // 8-byte store

    #pragma unroll
    for (int d = 16; d > 0; d >>= 1) s += __shfl_xor(s, d, 32);
    if (lane == 0) sq[row] = s;
}

// ---------------------------------------------------------------------------
// K2: k0_mean[i] = (1/N) * sum_j exp(-0.5*(sq[i]+sq[j]-2*dot(xi,xj)))
// Cooperative: 8 waves/block, one 16-row i-tile per wave. Each 16x128 B tile
// is staged once per block through double-buffered LDS (global traffic /8);
// next tile's global loads are issued before the current tile's WMMAs so the
// L2 latency hides behind compute, and inner-loop operands come from LDS.
// Diagonal forced to exactly 1.0 via branchless v_cndmask.
// ---------------------------------------------------------------------------
__global__ void k_rowmean(const bf16_t* __restrict__ xb,
                          const float* __restrict__ sq,
                          float* __restrict__ k0mean) {
    __shared__ __align__(16) char smem[2][16 * LDS_STRIDE];

    const int tid  = threadIdx.x;
    const int lane = tid & 31;
    const int wv   = tid >> 5;                    // wave in block: 0..7
    const int i0   = (blockIdx.x * 8 + wv) * 16;  // this wave's i-tile
    const int half = lane >> 4;                   // 0: lanes 0-15, 1: 16-31
    const int lr   = lane & 15;

    // staging role: 256 threads x 16B cover one 4KB tile
    const int srow = tid >> 4;                    // 0..15
    const int sseg = tid & 15;                    // 0..15

    // --- A fragments: 16x32 bf16 per K-chunk, contiguous K per lane half (same order as B) ---
    // lane half 0: K = {kb..kb+15} with kb = 32c
    // lane half 1: same with kb = 32c + 16
    v16bf a[4];
    {
        const bf16_t* arow = xb + (size_t)(i0 + lr) * DIMK;
        #pragma unroll
        for (int c = 0; c < 4; ++c) {
            const int kb = 32 * c + 16 * half;
            v8bf lo = *(const v8bf*)(arow + kb);
            v8bf hi = *(const v8bf*)(arow + kb + 8);
            #pragma unroll
            for (int e = 0; e < 8; ++e) { a[c][e] = lo[e]; a[c][8 + e] = hi[e]; }
        }
    }

    // sq for my 8 output rows: M = 8*half + v
    float sqi[8];
    {
        const float* p = sq + i0 + 8 * half;
        #pragma unroll
        for (int v = 0; v < 8; ++v) sqi[v] = p[v];
    }

    // diagonal hits when j0 + lr == i0 + 8*half + v  <=>  j0 == dbase + v
    const int dbase = i0 + 8 * half - lr;

    float accs[8];
    #pragma unroll
    for (int v = 0; v < 8; ++v) accs[v] = 0.0f;

    // ---- prologue: stage tile 0 into buffer 0 ----
    {
        float4 t0 = *(const float4*)(xb + (size_t)srow * DIMK + sseg * 8);
        *(float4*)(&smem[0][srow * LDS_STRIDE + sseg * 16]) = t0;
    }
    __syncthreads();

    int cur = 0;
    for (int j0 = 0; j0 < NROWS; j0 += 16) {
        // ---- issue next tile's global load (wraps on last iter; harmless) ----
        const int jn = (j0 + 16) & (NROWS - 1);
        const float4 pf = *(const float4*)(xb + (size_t)(jn + srow) * DIMK + sseg * 8);
        const float sqj = sq[j0 + lr];

        // ---- compute current tile from LDS ----
        // B 32x16 fragment cc: lane = column N=lr (row j0+lr of x),
        // 16 contiguous bf16 at K = 32*cc + 16*half -> byte 64*cc + 32*half.
        const char* bufr = &smem[cur][0];
        v8f c = {};
        #pragma unroll
        for (int cc = 0; cc < 4; ++cc) {
            v16bf b = *(const v16bf*)(bufr + lr * LDS_STRIDE + 64 * cc + 32 * half);
            c = __builtin_amdgcn_wmma_f32_16x16x32_bf16(
                    false, a[cc], false, b, (short)0, c, false, false);
        }

        #pragma unroll
        for (int v = 0; v < 8; ++v) {
            const float d2 = sqj + (sqi[v] - 2.0f * c[v]);
            const float e  = fast_exp2(NEG_HALF_LOG2E * d2);  // unconditional
            accs[v] += (j0 == dbase + v) ? 1.0f : e;          // v_cndmask
        }

        // ---- commit prefetched tile to the back buffer ----
        *(float4*)(&smem[cur ^ 1][srow * LDS_STRIDE + sseg * 16]) = pf;
        __syncthreads();   // back buffer ready; front buffer free for overwrite
        cur ^= 1;
    }

    // Reduce over the 16 lanes of each half (xor strides stay in-half).
    #pragma unroll
    for (int v = 0; v < 8; ++v) {
        float s = accs[v];
        #pragma unroll
        for (int d = 1; d < 16; d <<= 1) s += __shfl_xor(s, d, 32);
        accs[v] = s;
    }
    if (lr == 0) {
        float* outp = k0mean + i0 + 8 * half;
        #pragma unroll
        for (int v = 0; v < 8; ++v) outp[v] = accs[v] * (1.0f / NROWS);
    }
}

// ---------------------------------------------------------------------------
// K3: obj[i] = k0 - 2*k0_mean[i], with k0 = 1 for the RBF kernel.
// ---------------------------------------------------------------------------
__global__ void k_initobj(const float* __restrict__ k0m, float* __restrict__ obj) {
    const int i = blockIdx.x * blockDim.x + threadIdx.x;
    obj[i] = 1.0f - 2.0f * k0m[i];
}

// ---------------------------------------------------------------------------
// K4: obj[j] += 2*K(x[cur], x[j]) - 2*k0_mean[j]   (f32 dot, diagonal = 1)
// ---------------------------------------------------------------------------
__global__ void k_update(const float* __restrict__ x, const float* __restrict__ sq,
                         const float* __restrict__ k0m, const int* __restrict__ cur,
                         float* __restrict__ obj) {
    const int j = blockIdx.x * blockDim.x + threadIdx.x;
    const int c = *cur;
    const float4* xc = (const float4*)(x + (size_t)c * DIMK);
    const float4* xj = (const float4*)(x + (size_t)j * DIMK);
    float dot = 0.0f;
    #pragma unroll
    for (int k = 0; k < 32; ++k) {
        float4 av = xc[k], bv = xj[k];
        dot = fmaf(av.x, bv.x, fmaf(av.y, bv.y, fmaf(av.z, bv.z, fmaf(av.w, bv.w, dot))));
    }
    const float d2 = sq[c] + sq[j] - 2.0f * dot;
    const float e  = fast_exp2(NEG_HALF_LOG2E * d2);
    const float kv = (j == c) ? 1.0f : e;        // branchless
    obj[j] += 2.0f * kv - 2.0f * k0m[j];
}

// ---------------------------------------------------------------------------
// K5: single-block argmin with first-index tie-break (jnp.argmin semantics).
// Writes winner to out[t] (as float, per harness output dtype) and *cur.
// ---------------------------------------------------------------------------
__global__ void k_argmin(const float* __restrict__ obj, int* __restrict__ cur,
                         float* __restrict__ out, int t) {
    __shared__ float svals[1024];
    __shared__ int   sidx[1024];
    const int tid = threadIdx.x;
    float bv = 3.0e38f;
    int   bi = 0x7fffffff;
    for (int j = tid; j < NROWS; j += 1024) {
        const float v = obj[j];
        if (v < bv) { bv = v; bi = j; }   // strictly increasing j -> first min kept
    }
    svals[tid] = bv; sidx[tid] = bi;
    __syncthreads();
    for (int s = 512; s > 0; s >>= 1) {
        if (tid < s) {
            const float v2 = svals[tid + s];
            const int   i2 = sidx[tid + s];
            if (v2 < svals[tid] || (v2 == svals[tid] && i2 < sidx[tid])) {
                svals[tid] = v2; sidx[tid] = i2;
            }
        }
        __syncthreads();
    }
    if (tid == 0) {
        *cur   = sidx[0];
        out[t] = (float)sidx[0];
    }
}

// ---------------------------------------------------------------------------
extern "C" void kernel_launch(void* const* d_in, const int* in_sizes, int n_in,
                              void* d_out, int out_size, void* d_ws, size_t ws_size,
                              hipStream_t stream) {
    const float* x = (const float*)d_in[0];
    const int m = out_size;                 // reference m == number of outputs
    float* out = (float*)d_out;

    char* ws = (char*)d_ws;
    bf16_t* xb  = (bf16_t*)ws;                                   // 4 MiB
    float*  sq  = (float*)(ws + (size_t)NROWS * DIMK * 2);       // 64 KiB
    float*  k0m = sq  + NROWS;                                   // 64 KiB
    float*  obj = k0m + NROWS;                                   // 64 KiB
    int*    cur = (int*)(obj + NROWS);

    // 1) norms + bf16 copy: 8 waves/block, one wave per row
    k_prep<<<NROWS / 8, 256, 0, stream>>>(x, xb, sq);
    // 2) Gram row means via WMMA: 1024 i-tiles, 8 waves/block, LDS-staged B
    k_rowmean<<<NROWS / 16 / 8, 256, 0, stream>>>(xb, sq, k0m);
    // 3) objective init + first selection
    k_initobj<<<NROWS / 256, 256, 0, stream>>>(k0m, obj);
    k_argmin<<<1, 1024, 0, stream>>>(obj, cur, out, 0);
    // 4) greedy thinning loop
    for (int t = 1; t < m; ++t) {
        k_update<<<NROWS / 256, 256, 0, stream>>>(x, sq, k0m, cur, obj);
        k_argmin<<<1, 1024, 0, stream>>>(obj, cur, out, t);
    }
}
